// Cross_Attention_66511863545888
// MI455X (gfx1250) — matrix operand order repaired and checked by reading, not compiled, not run on hardware
//
#include <hip/hip_runtime.h>
#include <hip/hip_bf16.h>
#include <math.h>

#define N_POS 4096
#define CDIM 128
#define HEADS 4
#define HDIM 32
#define BATCH 2

typedef __attribute__((ext_vector_type(16))) __bf16       bf16x16;
typedef __attribute__((ext_vector_type(8)))  float        f32x8;
typedef __attribute__((ext_vector_type(4)))  unsigned int u32x4;

union FragBF {
    bf16x16        f;
    unsigned short u[16];
    unsigned int   w[8];
    u32x4          q[2];
};

__device__ __forceinline__ unsigned short f32_to_bf16(float x) {
    unsigned int u = __float_as_uint(x);
    u += 0x7fffu + ((u >> 16) & 1u);   // round-to-nearest-even
    return (unsigned short)(u >> 16);
}

__device__ __forceinline__ unsigned int pack_bf16x2(float lo, float hi) {
    return (unsigned int)f32_to_bf16(lo) | ((unsigned int)f32_to_bf16(hi) << 16);
}

// Raw v_exp_f32: our arguments are always <= 0, so no range fixup is needed;
// deep-negative underflow flushes to 0, which is exactly the softmax tail.
__device__ __forceinline__ float exp2_raw(float x) {
    return __builtin_amdgcn_exp2f(x);
}

// Rebuild a 16x32 bf16 A-fragment from LDS words that hold interleaved pairs
// (K=k in low half, K=k+16 in high half) at word index k (k = 0..15 per row).
// Lane reads words [koff .. koff+7]; fragment wants K {koff..koff+7} in
// elements 0..7 and K {koff+16..koff+23} in elements 8..15.
__device__ __forceinline__ void deinterleave_frag(FragBF& pa, const unsigned int* pr) {
    u32x4 a = *(const u32x4*)pr;
    u32x4 b = *(const u32x4*)(pr + 4);
    unsigned int w0[8] = {a[0], a[1], a[2], a[3], b[0], b[1], b[2], b[3]};
#pragma unroll
    for (int j = 0; j < 4; ++j) {
        pa.w[j]     = (w0[2 * j] & 0xffffu) | (w0[2 * j + 1] << 16);
        pa.w[4 + j] = (w0[2 * j] >> 16)     | (w0[2 * j + 1] & 0xffff0000u);
    }
}

// -----------------------------------------------------------------------------
// Kernel 1: Q/K/V 1x1-conv GEMM (f32), fused L2 normalize over the SPATIAL axis
// (per (b, channel) row of length N, matching the reference), emit bf16.
// SCALE*log2(e) is folded into Q so the attention hot loop runs the whole
// softmax in the log2 domain with zero per-element scale multiplies.
//   Q,K -> [b*h][n][d] bf16 (row = position, d contiguous)
//   V   -> [b*h][d][n] bf16 (transposed: d-major)
// grid = (256 rows = b*128, 3 matrices), block = 256 threads, 16 n per thread.
// -----------------------------------------------------------------------------
__global__ __launch_bounds__(256) void qkv_norm_kernel(
    const float* __restrict__ x, const float* __restrict__ cond,
    const float* __restrict__ Wq, const float* __restrict__ bq,
    const float* __restrict__ Wk, const float* __restrict__ bk,
    const float* __restrict__ Wv, const float* __restrict__ bv,
    unsigned short* __restrict__ Qbf, unsigned short* __restrict__ Kbf,
    unsigned short* __restrict__ Vt)
{
    const int row   = blockIdx.x;      // b*128 + o
    const int which = blockIdx.y;      // 0=Q, 1=K, 2=V
    const int b = row >> 7;
    const int o = row & 127;
    const int t = threadIdx.x;         // 0..255

    __shared__ float wrow[CDIM];
    __shared__ float red[256];

    const float* W    = (which == 0) ? Wq : (which == 1) ? Wk : Wv;
    const float* bias = (which == 0) ? bq : (which == 1) ? bk : bv;
    const float* X    = (which == 0) ? x  : cond;

    if (t < CDIM) wrow[t] = W[o * CDIM + t];
    __syncthreads();

    float acc[16];
#pragma unroll
    for (int i = 0; i < 16; ++i) acc[i] = bias[o];

    const float* Xb = X + (size_t)b * CDIM * N_POS;
    for (int c = 0; c < CDIM; ++c) {
        float wc = wrow[c];
        const float* xr = Xb + (size_t)c * N_POS + t;
#pragma unroll
        for (int i = 0; i < 16; ++i) acc[i] = fmaf(wc, xr[i * 256], acc[i]);
    }

    float scale = 1.0f;
    if (which < 2) {
        float ss = 0.0f;
#pragma unroll
        for (int i = 0; i < 16; ++i) ss = fmaf(acc[i], acc[i], ss);
        red[t] = ss;
        __syncthreads();
        for (int s = 128; s > 0; s >>= 1) {
            if (t < s) red[t] += red[t + s];
            __syncthreads();
        }
        float nrm = sqrtf(red[0]);
        // Fold SCALE (=10) * log2(e) into Q only: sim arrives in log2 domain.
        scale = ((which == 0) ? 10.0f * 1.44269504088896340736f : 1.0f)
                / fmaxf(nrm, 1e-12f);
    }

    const int h  = o >> 5;
    const int d  = o & 31;
    const int bh = b * HEADS + h;

    if (which == 2) {
        unsigned short* dst = Vt + ((size_t)(bh * HDIM + d)) * N_POS + t;
#pragma unroll
        for (int i = 0; i < 16; ++i) dst[i * 256] = f32_to_bf16(acc[i]);
    } else {
        unsigned short* dst = ((which == 0) ? Qbf : Kbf) + (size_t)bh * N_POS * HDIM + d;
#pragma unroll
        for (int i = 0; i < 16; ++i)
            dst[(size_t)(t + i * 256) * HDIM] = f32_to_bf16(acc[i] * scale);
    }
}

// -----------------------------------------------------------------------------
// Kernel 2: flash attention, one wave per 16-query tile, 64 keys per block.
//  - 4x V_WMMA_F32_16X16X32_BF16 for Q·K^T (sim already in log2 domain)
//  - row max via 4-stage half-wave shuffles (rows of the C/D tile live in
//    16-lane half groups at vgpr index r), amortized over 64 columns
//  - exp2 via raw v_exp_f32 (args always <= 0, underflow->0 is correct)
//  - P staged through LDS as packed bf16 pairs (stride-20 u32 rows: the two
//    half-wave groups hit disjoint bank sets)
//  - row SUMS via WMMA against an all-ones B fragment (D tile of P x 1 has
//    exactly the rsum[] register layout) -> no sum shuffles at all
//  - 4x WMMA for P·V with f32 accumulate
// Output: attn_out channel-major f32 [b][c=128][n].
// -----------------------------------------------------------------------------
__global__ __launch_bounds__(128) void flash_attn_kernel(
    const unsigned short* __restrict__ Qbf,
    const unsigned short* __restrict__ Kbf,
    const unsigned short* __restrict__ Vt,
    float* __restrict__ attn_out)
{
    __shared__ unsigned int PldsA[4][16 * 20];
    __shared__ unsigned int PldsB[4][16 * 20];

    const int lane = threadIdx.x & 31;
    const int wid  = threadIdx.x >> 5;
    const int tile = blockIdx.x * 4 + wid;      // 0..2047
    const int qt   = tile & 255;                // query tile within (b,h)
    const int bh   = tile >> 8;                 // 0..7
    const int b    = bh >> 2;
    const int h    = bh & 3;
    const int n0   = qt * 16;
    const int col  = lane & 15;
    const int hl   = lane >> 4;
    const int koff = hl * 8;

    // Q A-fragment (16 queries x K=32 head dims); SCALE*log2e already folded.
    FragBF qa;
    {
        const unsigned short* qrow = Qbf + ((size_t)bh * N_POS + n0 + col) * HDIM;
        qa.q[0] = *(const u32x4*)(qrow + 2 * koff);
        qa.q[1] = *(const u32x4*)(qrow + 2 * koff + 8);
    }

    // All-ones B fragment for WMMA row sums (layout-independent).
    FragBF ones;
#pragma unroll
    for (int j = 0; j < 8; ++j) ones.w[j] = 0x3F803F80u;

    float rmax[8], rsum[8];
    f32x8 O0 = {0.f, 0.f, 0.f, 0.f, 0.f, 0.f, 0.f, 0.f};
    f32x8 O1 = {0.f, 0.f, 0.f, 0.f, 0.f, 0.f, 0.f, 0.f};
#pragma unroll
    for (int r = 0; r < 8; ++r) { rmax[r] = -1.0e30f; rsum[r] = 0.0f; }

    const unsigned short* Kbase = Kbf + (size_t)bh * N_POS * HDIM;
    const unsigned short* Vbase = Vt + (size_t)bh * HDIM * N_POS;
    const f32x8 zc = {0.f, 0.f, 0.f, 0.f, 0.f, 0.f, 0.f, 0.f};

    for (int j0 = 0; j0 < N_POS; j0 += 64) {
        // ---- Q.K^T for 64 keys: 4 B-fragments, 4 WMMAs -------------------
        const unsigned short* kp = Kbase + (size_t)(j0 + col) * HDIM + hl * 16;
        FragBF kb0, kb1, kb2, kb3;
        kb0.q[0] = ((const u32x4*)(kp            ))[0]; kb0.q[1] = ((const u32x4*)(kp            ))[1];
        kb1.q[0] = ((const u32x4*)(kp + 16*HDIM  ))[0]; kb1.q[1] = ((const u32x4*)(kp + 16*HDIM  ))[1];
        kb2.q[0] = ((const u32x4*)(kp + 32*HDIM  ))[0]; kb2.q[1] = ((const u32x4*)(kp + 32*HDIM  ))[1];
        kb3.q[0] = ((const u32x4*)(kp + 48*HDIM  ))[0]; kb3.q[1] = ((const u32x4*)(kp + 48*HDIM  ))[1];

        f32x8 s0 = __builtin_amdgcn_wmma_f32_16x16x32_bf16(false, qa.f, false, kb0.f, (short)0, zc, false, false);
        f32x8 s1 = __builtin_amdgcn_wmma_f32_16x16x32_bf16(false, qa.f, false, kb1.f, (short)0, zc, false, false);
        f32x8 s2 = __builtin_amdgcn_wmma_f32_16x16x32_bf16(false, qa.f, false, kb2.f, (short)0, zc, false, false);
        f32x8 s3 = __builtin_amdgcn_wmma_f32_16x16x32_bf16(false, qa.f, false, kb3.f, (short)0, zc, false, false);

        // ---- online softmax over the 64-column stripe (log2 domain) ------
        float corr[8];
#pragma unroll
        for (int r = 0; r < 8; ++r) {
            float a0 = s0[r], a1 = s1[r], a2 = s2[r], a3 = s3[r];
            float tmx = fmaxf(fmaxf(a0, a1), fmaxf(a2, a3));
            tmx = fmaxf(tmx, __shfl_xor(tmx, 1, 32));
            tmx = fmaxf(tmx, __shfl_xor(tmx, 2, 32));
            tmx = fmaxf(tmx, __shfl_xor(tmx, 4, 32));
            tmx = fmaxf(tmx, __shfl_xor(tmx, 8, 32));
            float nm = fmaxf(rmax[r], tmx);
            corr[r]  = exp2_raw(rmax[r] - nm);
            rmax[r]  = nm;
            float p0 = exp2_raw(a0 - nm);
            float p1 = exp2_raw(a1 - nm);
            float p2 = exp2_raw(a2 - nm);
            float p3 = exp2_raw(a3 - nm);
            const int idx = (r + 8 * hl) * 20 + col;     // pad 20: halves conflict-free
            PldsA[wid][idx] = pack_bf16x2(p0, p1);       // keys (col, col+16)
            PldsB[wid][idx] = pack_bf16x2(p2, p3);       // keys (32+col, 48+col)
            O0[r] *= corr[r];
            O1[r] *= corr[r];
        }

        // Drain DS writes before the same wave re-reads P in A-layout.
        asm volatile("s_wait_dscnt 0" ::: "memory");

        FragBF pa0, pa1;
        deinterleave_frag(pa0, &PldsA[wid][col * 20 + koff]);
        deinterleave_frag(pa1, &PldsB[wid][col * 20 + koff]);

        // ---- V B-fragments (transposed V: d column per lane, 16 keys) ----
        const unsigned short* vp = Vbase + (size_t)col * N_POS + j0 + hl * 8;
        FragBF vb0, vb1, vb2, vb3;
        vb0.q[0] = ((const u32x4*)(vp                   ))[0]; vb0.q[1] = ((const u32x4*)(vp                   ))[2];
        vb1.q[0] = ((const u32x4*)(vp + (size_t)16*N_POS))[0]; vb1.q[1] = ((const u32x4*)(vp + (size_t)16*N_POS))[2];
        vb2.q[0] = ((const u32x4*)(vp + 32                   ))[0]; vb2.q[1] = ((const u32x4*)(vp + 32                   ))[2];
        vb3.q[0] = ((const u32x4*)(vp + (size_t)16*N_POS + 32))[0]; vb3.q[1] = ((const u32x4*)(vp + (size_t)16*N_POS + 32))[2];

        // ---- P.V accumulation + WMMA row sums ----------------------------
        O0 = __builtin_amdgcn_wmma_f32_16x16x32_bf16(false, pa0.f, false, vb0.f, (short)0, O0, false, false);
        O1 = __builtin_amdgcn_wmma_f32_16x16x32_bf16(false, pa0.f, false, vb1.f, (short)0, O1, false, false);
        O0 = __builtin_amdgcn_wmma_f32_16x16x32_bf16(false, pa1.f, false, vb2.f, (short)0, O0, false, false);
        O1 = __builtin_amdgcn_wmma_f32_16x16x32_bf16(false, pa1.f, false, vb3.f, (short)0, O1, false, false);

        f32x8 ts = __builtin_amdgcn_wmma_f32_16x16x32_bf16(false, pa0.f, false, ones.f, (short)0, zc, false, false);
        ts       = __builtin_amdgcn_wmma_f32_16x16x32_bf16(false, pa1.f, false, ones.f, (short)0, ts, false, false);
#pragma unroll
        for (int r = 0; r < 8; ++r) rsum[r] = rsum[r] * corr[r] + ts[r];
    }

    // Epilogue: divide by row sums, scatter to channel-major attn_out[b][c][n].
#pragma unroll
    for (int r = 0; r < 8; ++r) {
        const float inv = 1.0f / rsum[r];
        const int m = n0 + r + 8 * hl;
        attn_out[((size_t)(b * CDIM + h * HDIM + col)) * N_POS + m]      = O0[r] * inv;
        attn_out[((size_t)(b * CDIM + h * HDIM + 16 + col)) * N_POS + m] = O1[r] * inv;
    }
}

// -----------------------------------------------------------------------------
// Kernel 3: output projection (f32 VALU for final accuracy).
// y[b][o][n] = sum_c Wo[o][c] * attn_out[b][c][n] + bo[o]
// grid = (256 rows, 16 n-chunks), block = 256.
// -----------------------------------------------------------------------------
__global__ __launch_bounds__(256) void proj_kernel(
    const float* __restrict__ attn_out,
    const float* __restrict__ Wo, const float* __restrict__ bo,
    float* __restrict__ out)
{
    const int row = blockIdx.x;          // b*128 + o
    const int b = row >> 7;
    const int o = row & 127;
    const int n = blockIdx.y * 256 + threadIdx.x;

    __shared__ float wrow[CDIM];
    if (threadIdx.x < CDIM) wrow[threadIdx.x] = Wo[o * CDIM + threadIdx.x];
    __syncthreads();

    float acc = bo[o];
    const float* A = attn_out + (size_t)b * CDIM * N_POS + n;
#pragma unroll 4
    for (int c = 0; c < CDIM; ++c) acc = fmaf(wrow[c], A[(size_t)c * N_POS], acc);

    out[(size_t)row * N_POS + n] = acc;
}

// -----------------------------------------------------------------------------
extern "C" void kernel_launch(void* const* d_in, const int* in_sizes, int n_in,
                              void* d_out, int out_size, void* d_ws, size_t ws_size,
                              hipStream_t stream) {
    const float* x      = (const float*)d_in[0];
    const float* cond_x = (const float*)d_in[1];
    const float* Wq = (const float*)d_in[2];
    const float* bq = (const float*)d_in[3];
    const float* Wk = (const float*)d_in[4];
    const float* bk = (const float*)d_in[5];
    const float* Wv = (const float*)d_in[6];
    const float* bv = (const float*)d_in[7];
    const float* Wo = (const float*)d_in[8];
    const float* bo = (const float*)d_in[9];
    float* out = (float*)d_out;

    // Workspace carve-up (10 MB total):
    //   Qbf, Kbf : [8][4096][32] bf16 (2 MB each)
    //   Vt       : [8][32][4096] bf16 (2 MB)
    //   attn_out : [2][128][4096] f32 (4 MB)
    const size_t ELEMS = (size_t)BATCH * HEADS * N_POS * HDIM;   // 1,048,576
    unsigned short* Qbf = (unsigned short*)d_ws;
    unsigned short* Kbf = Qbf + ELEMS;
    unsigned short* Vt  = Kbf + ELEMS;
    float* attn = (float*)(Vt + ELEMS);

    qkv_norm_kernel<<<dim3(BATCH * CDIM, 3), 256, 0, stream>>>(
        x, cond_x, Wq, bq, Wk, bk, Wv, bv, Qbf, Kbf, Vt);

    // 2048 query tiles (b*h*256), 4 waves per block of 128 threads.
    flash_attn_kernel<<<512, 128, 0, stream>>>(Qbf, Kbf, Vt, attn);

    proj_kernel<<<dim3(BATCH * CDIM, N_POS / 256), 256, 0, stream>>>(
        attn, Wo, bo, out);
}
